// IDKN_10531259809801
// MI455X (gfx1250) — compile-verified, matrix operand order checked by reading, not run on hardware
//
#include <hip/hip_runtime.h>
#include <hip/hip_bf16.h>

// ---------------------------------------------------------------------------
// GAT pipeline for MI455X (gfx1250), wave32 + WMMA.
//   3x [ h = x@W (branchless v_wmma_f32_16x16x32_f16, f16 staged inputs) ;
//        per-head edge softmax ; scatter-aggregate ; bias+relu ]
//   then y = adj@x6 via deduplicated sparse edges (bitmask),
//   out[i] = x6[i].y[i] + x1[i,0] + y[i].lin2_w
// ---------------------------------------------------------------------------

#define NN 10000          // 625 * 16 exactly -> no row bound checks in GEMM
#define NE 320000
#define ROWTILES 625
#define WAVES_PER_COL 80  // persistent waves per column tile

typedef __attribute__((ext_vector_type(16))) _Float16 v16h;
typedef __attribute__((ext_vector_type(8)))  _Float16 v8h;
typedef __attribute__((ext_vector_type(8)))  float    v8f;

// ---------------- staging: pad/convert x1 [n,10] -> Xh [n,32] f16 ----------
__global__ void pad_x1_kernel(const float* __restrict__ x1, _Float16* __restrict__ Xh, int n) {
    int t = blockIdx.x * blockDim.x + threadIdx.x;
    if (t >= n * 32) return;
    int row = t >> 5, k = t & 31;
    Xh[t] = (k < 10) ? (_Float16)x1[row * 10 + k] : (_Float16)0.0f;
}

// ---------------- staging: pack W into B-operand lane layout ---------------
// Bpack[(ct*32 + lane)*16 + e] = f16(W[k(e,half)][ct*16 + (lane&15)]), 0-padded
__global__ void pack_w_kernel(const float* __restrict__ W, _Float16* __restrict__ Bpack,
                              int in_c, int out_c, int ncol) {
    int t = blockIdx.x * blockDim.x + threadIdx.x;
    if (t >= ncol * 32 * 16) return;
    int e    = t & 15;
    int lane = (t >> 4) & 31;
    int ct   = t >> 9;
    int half = lane >> 4;
    int m    = lane & 15;
    int k    = (e < 8 ? e : e + 8) + 8 * half;
    int colN = ct * 16 + m;
    float v = (k < in_c && colN < out_c) ? W[k * out_c + colN] : 0.0f;
    Bpack[t] = (_Float16)v;
}

// ---------------- WMMA GEMM: H[n(row-stride hs)] = Xh[n,32] @ Bpack --------
// Branchless: A = 2x b128 loads, B = 2x b128 loads (once per wave),
// 1 v_wmma_f32_16x16x32_f16 per row tile, 8 unconditional b32 stores.
__global__ void wmma_gemm_kernel(const _Float16* __restrict__ Xh,
                                 const _Float16* __restrict__ Bpack,
                                 float* __restrict__ H, int ncol, int hs) {
    int gwave = blockIdx.x * (blockDim.x >> 5) + (threadIdx.x >> 5);
    int ct = gwave / WAVES_PER_COL;
    int wv = gwave % WAVES_PER_COL;
    if (ct >= ncol) return;                          // wave-uniform
    int lane = threadIdx.x & 31;
    int half = lane >> 4;
    int colN = ct * 16 + (lane & 15);

    const _Float16* bp = Bpack + ((size_t)(ct * 32 + lane) << 4);
    v8h blo = *(const v8h*)(bp);
    v8h bhi = *(const v8h*)(bp + 8);
    v16h b = __builtin_shufflevector(blo, bhi, 0,1,2,3,4,5,6,7,8,9,10,11,12,13,14,15);

    for (int rt = wv; rt < ROWTILES; rt += WAVES_PER_COL) {
        int row = rt * 16 + (lane & 15);
        const _Float16* xp = Xh + ((size_t)row << 5) + 8 * half;
        __builtin_prefetch(Xh + (((size_t)(row + 16 * WAVES_PER_COL)) << 5), 0, 1);
        v8h alo = *(const v8h*)(xp);                 // K run {0..7}+8h
        v8h ahi = *(const v8h*)(xp + 16);            // K run {16..23}+8h
        v16h a = __builtin_shufflevector(alo, ahi, 0,1,2,3,4,5,6,7,8,9,10,11,12,13,14,15);
        v8f c = {};
        v8f d = __builtin_amdgcn_wmma_f32_16x16x32_f16(
            false, a, false, b, (short)0, c, false, false);
        float* hp = H + (size_t)(rt * 16) * hs + colN;
        #pragma unroll
        for (int e = 0; e < 8; ++e)                  // D: VGPR e -> M = e + 8*half
            hp[(size_t)(e + 8 * half) * hs] = d[e];
    }
}

// ---------------- attention scalars: e_src/e_dst = <h, a_src/a_dst> --------
__global__ void eh_kernel(const float* __restrict__ H,
                          const float* __restrict__ asr,
                          const float* __restrict__ ads,
                          float* __restrict__ esrc, float* __restrict__ edst,
                          int n, int heads, int C, int hs) {
    int t = blockIdx.x * blockDim.x + threadIdx.x;
    if (t >= n * heads) return;
    int i = t / heads, hd = t % heads;
    const float* hp = H + (size_t)i * hs + hd * C;
    float s1 = 0.f, s2 = 0.f;
    for (int c = 0; c < C; ++c) {
        float v = hp[c];
        s1 += v * asr[hd * C + c];
        s2 += v * ads[hd * C + c];
    }
    esrc[t] = s1;
    edst[t] = s2;
}

// order-preserving float<->uint key for atomic max
__device__ __forceinline__ unsigned fkey(float f) {
    unsigned u = __float_as_uint(f);
    return (u & 0x80000000u) ? ~u : (u | 0x80000000u);
}
__device__ __forceinline__ float funkey(unsigned k) {
    unsigned u = (k & 0x80000000u) ? (k & 0x7fffffffu) : ~k;
    return __uint_as_float(u);
}

__device__ __forceinline__ void edge_sd(const int* row, const int* col,
                                        int e, int E, int& s, int& d) {
    if (e < E) { s = row[e]; d = col[e]; }
    else       { s = e - E; d = e - E; }             // appended self-loops
}

__device__ __forceinline__ float leaky(float v) { return v > 0.f ? v : 0.2f * v; }

// pass 1: segment max (uint keys, init 0 == below all real values)
__global__ void edge_max_kernel(const int* __restrict__ row, const int* __restrict__ col,
                                const float* __restrict__ esrc, const float* __restrict__ edst,
                                unsigned* __restrict__ segmax, int E, int n, int heads) {
    int t = blockIdx.x * blockDim.x + threadIdx.x;
    if (t >= (E + n) * heads) return;
    int e = t / heads, hd = t % heads;
    int s, d; edge_sd(row, col, e, E, s, d);
    float v = leaky(esrc[s * heads + hd] + edst[d * heads + hd]);
    atomicMax(&segmax[d * heads + hd], fkey(v));
}

// pass 2: segment sum of exp(v - max)
__global__ void edge_sum_kernel(const int* __restrict__ row, const int* __restrict__ col,
                                const float* __restrict__ esrc, const float* __restrict__ edst,
                                const unsigned* __restrict__ segmax,
                                float* __restrict__ segsum, int E, int n, int heads) {
    int t = blockIdx.x * blockDim.x + threadIdx.x;
    if (t >= (E + n) * heads) return;
    int e = t / heads, hd = t % heads;
    int s, d; edge_sd(row, col, e, E, s, d);
    float v  = leaky(esrc[s * heads + hd] + edst[d * heads + hd]);
    float mx = funkey(segmax[d * heads + hd]);
    atomicAdd(&segsum[d * heads + hd], __expf(v - mx));
}

// pass 3: acc[d] += alpha * h[s]
__global__ void edge_acc_kernel(const int* __restrict__ row, const int* __restrict__ col,
                                const float* __restrict__ esrc, const float* __restrict__ edst,
                                const unsigned* __restrict__ segmax,
                                const float* __restrict__ segsum,
                                const float* __restrict__ H,
                                float* __restrict__ acc, int E, int n, int heads, int C, int hs) {
    int t = blockIdx.x * blockDim.x + threadIdx.x;
    if (t >= (E + n) * heads) return;
    int e = t / heads, hd = t % heads;
    int s, d; edge_sd(row, col, e, E, s, d);
    float v  = leaky(esrc[s * heads + hd] + edst[d * heads + hd]);
    float mx = funkey(segmax[d * heads + hd]);
    float alpha = __expf(v - mx) / segsum[d * heads + hd];
    const float* hp = H + (size_t)s * hs + hd * C;
    float* ad = acc + (size_t)d * (heads * C) + hd * C;
    for (int c = 0; c < C; ++c) atomicAdd(&ad[c], alpha * hp[c]);
}

// bias + relu; also write next layer's f16 zero-padded A staging
__global__ void bias_relu_kernel(const float* __restrict__ acc,
                                 const float* __restrict__ bias,
                                 float* __restrict__ xn, _Float16* __restrict__ Xh,
                                 int n, int out) {
    int t = blockIdx.x * blockDim.x + threadIdx.x;
    if (t >= n * 32) return;
    int row = t >> 5, k = t & 31;
    float v = 0.0f;
    if (k < out) {
        v = acc[row * out + k] + bias[k];
        v = v > 0.f ? v : 0.f;
        xn[row * out + k] = v;
    }
    Xh[t] = (_Float16)v;
}

// ---------------- adjacency (deduplicated, row-normalized) -----------------
__global__ void deg_init_kernel(float* __restrict__ deg, int n) {
    int i = blockIdx.x * blockDim.x + threadIdx.x;
    if (i < n) deg[i] = 1.0f;                        // diagonal contribution
}

__global__ void dedup_kernel(const int* __restrict__ row, const int* __restrict__ col,
                             unsigned* __restrict__ mask, float* __restrict__ deg,
                             unsigned char* __restrict__ rep, int E, int n, int use_mask) {
    int e = blockIdx.x * blockDim.x + threadIdx.x;
    if (e >= E) return;
    int i = row[e], j = col[e];
    if (i == j) { rep[e] = 0; return; }              // diagonal already counted
    if (use_mask) {
        long long bit = (long long)i * n + j;
        unsigned b = 1u << (bit & 31);
        unsigned old = atomicOr(&mask[bit >> 5], b);
        if (!(old & b)) { rep[e] = 1; atomicAdd(&deg[i], 1.0f); }
        else rep[e] = 0;
    } else {
        rep[e] = 1; atomicAdd(&deg[i], 1.0f);        // fallback: no dedup
    }
}

__global__ void y_init_kernel(const float* __restrict__ x6, float* __restrict__ y, int n) {
    int t = blockIdx.x * blockDim.x + threadIdx.x;
    if (t < n * 8) y[t] = x6[t];                     // diagonal term of adj@x6
}

__global__ void y_edge_kernel(const int* __restrict__ row, const int* __restrict__ col,
                              const unsigned char* __restrict__ rep,
                              const float* __restrict__ x6, float* __restrict__ y, int E) {
    int t = blockIdx.x * blockDim.x + threadIdx.x;
    if (t >= E * 8) return;
    int e = t >> 3, c = t & 7;
    if (!rep[e]) return;
    int i = row[e], j = col[e];
    atomicAdd(&y[i * 8 + c], x6[j * 8 + c]);
}

__global__ void final_kernel(const float* __restrict__ x1, const float* __restrict__ x6,
                             const float* __restrict__ y, const float* __restrict__ deg,
                             const float* __restrict__ lw, float* __restrict__ out, int n) {
    int i = blockIdx.x * blockDim.x + threadIdx.x;
    if (i >= n) return;
    float inv = 1.0f / fmaxf(deg[i], 1e-12f);
    float r2 = 0.f, gs = 0.f;
    for (int c = 0; c < 8; ++c) {
        float yv = y[i * 8 + c] * inv;               // (adj@x6)[i,c]
        r2 += x6[i * 8 + c] * yv;
        gs += yv * lw[c];
    }
    out[i] = r2 + x1[i * 10] + gs;
}

// ---------------------------------------------------------------------------
static void run_gat_layer(const _Float16* Xh, int in_c, int heads, int C,
                          const float* W, const float* asr, const float* ads,
                          const float* bias, const int* row, const int* col,
                          _Float16* Bpack, float* h, float* acc,
                          float* esrc, float* edst, unsigned* segmax, float* segsum,
                          float* xnext, _Float16* XhNext, hipStream_t stream) {
    const int n = NN, E = NE, out = heads * C;
    const int ncol = (out + 15) / 16;
    const int hs   = 16 * ncol;                      // padded H row stride

    hipMemsetAsync(segmax, 0, (size_t)n * heads * sizeof(unsigned), stream);
    hipMemsetAsync(segsum, 0, (size_t)n * heads * sizeof(float), stream);
    hipMemsetAsync(acc,    0, (size_t)n * out * sizeof(float), stream);

    pack_w_kernel<<<(ncol * 32 * 16 + 255) / 256, 256, 0, stream>>>(W, Bpack, in_c, out, ncol);
    // ncol * WAVES_PER_COL waves, 8 waves per 256-thread block
    wmma_gemm_kernel<<<(ncol * WAVES_PER_COL + 7) / 8, 256, 0, stream>>>(Xh, Bpack, h, ncol, hs);

    eh_kernel<<<(n * heads + 255) / 256, 256, 0, stream>>>(h, asr, ads, esrc, edst, n, heads, C, hs);

    int tot = (E + n) * heads;
    int blk = (tot + 255) / 256;
    edge_max_kernel<<<blk, 256, 0, stream>>>(row, col, esrc, edst, segmax, E, n, heads);
    edge_sum_kernel<<<blk, 256, 0, stream>>>(row, col, esrc, edst, segmax, segsum, E, n, heads);
    edge_acc_kernel<<<blk, 256, 0, stream>>>(row, col, esrc, edst, segmax, segsum, h, acc,
                                             E, n, heads, C, hs);
    bias_relu_kernel<<<(n * 32 + 255) / 256, 256, 0, stream>>>(acc, bias, xnext, XhNext, n, out);
}

extern "C" void kernel_launch(void* const* d_in, const int* in_sizes, int n_in,
                              void* d_out, int out_size, void* d_ws, size_t ws_size,
                              hipStream_t stream) {
    const float* x1    = (const float*)d_in[0];
    const int*   ei    = (const int*)  d_in[2];      // [2, NE]
    const float* W1    = (const float*)d_in[4];
    const float* asr1  = (const float*)d_in[5];
    const float* ads1  = (const float*)d_in[6];
    const float* b1    = (const float*)d_in[7];
    const float* W2    = (const float*)d_in[8];
    const float* asr2  = (const float*)d_in[9];
    const float* ads2  = (const float*)d_in[10];
    const float* b2    = (const float*)d_in[11];
    const float* W3    = (const float*)d_in[12];
    const float* asr3  = (const float*)d_in[13];
    const float* ads3  = (const float*)d_in[14];
    const float* b3    = (const float*)d_in[15];
    const float* lin2w = (const float*)d_in[16];
    float* out = (float*)d_out;

    const int* row = ei;              // edge_index[0]
    const int* col = ei + NE;         // edge_index[1]

    // workspace layout (every buffer fully written before any read)
    char* ws = (char*)d_ws;
    size_t off = 0;
    auto take = [&](size_t bytes) {
        char* p = ws + off; off = (off + bytes + 255) & ~(size_t)255; return p;
    };
    float*     h      = (float*)    take((size_t)NN * 32 * 4);
    float*     acc    = (float*)    take((size_t)NN * 32 * 4);
    float*     xcur   = (float*)    take((size_t)NN * 32 * 4);
    _Float16*  Xh     = (_Float16*) take((size_t)NN * 32 * 2 + 64 * WAVES_PER_COL * 32 * 2); // + prefetch slack
    _Float16*  Bpack  = (_Float16*) take((size_t)2 * 32 * 16 * 2);
    float*     esrc   = (float*)    take((size_t)NN * 8 * 4);
    float*     edst   = (float*)    take((size_t)NN * 8 * 4);
    unsigned*  segmax = (unsigned*) take((size_t)NN * 8 * 4);
    float*     segsum = (float*)    take((size_t)NN * 8 * 4);
    float*     deg    = (float*)    take((size_t)NN * 4);
    float*     y      = (float*)    take((size_t)NN * 8 * 4);
    unsigned char* rep = (unsigned char*)take((size_t)NE);
    size_t mask_words = ((size_t)NN * NN + 31) / 32;
    size_t need_mask  = off + mask_words * 4;
    int use_mask = (ws_size >= need_mask) ? 1 : 0;
    unsigned* mask = use_mask ? (unsigned*)take(mask_words * 4) : (unsigned*)nullptr;

    // --- three GAT layers (x1 -> x3 -> x4 -> x6; activations in xcur) ---
    pad_x1_kernel<<<(NN * 32 + 255) / 256, 256, 0, stream>>>(x1, Xh, NN);
    run_gat_layer(Xh, 10, 8, 4, W1, asr1, ads1, b1, row, col, Bpack,
                  h, acc, esrc, edst, segmax, segsum, xcur, Xh, stream);
    run_gat_layer(Xh, 32, 4, 4, W2, asr2, ads2, b2, row, col, Bpack,
                  h, acc, esrc, edst, segmax, segsum, xcur, Xh, stream);
    run_gat_layer(Xh, 16, 2, 4, W3, asr3, ads3, b3, row, col, Bpack,
                  h, acc, esrc, edst, segmax, segsum, xcur, Xh, stream);
    const float* x6 = xcur;           // [NN, 8]

    // --- y = adj @ x6 (dedup distinct edges + diagonal, row-normalized) ---
    if (use_mask)
        hipMemsetAsync(mask, 0, mask_words * 4, stream);
    deg_init_kernel<<<(NN + 255) / 256, 256, 0, stream>>>(deg, NN);
    dedup_kernel<<<(NE + 255) / 256, 256, 0, stream>>>(row, col, mask, deg, rep, NE, NN, use_mask);
    y_init_kernel<<<(NN * 8 + 255) / 256, 256, 0, stream>>>(x6, y, NN);
    y_edge_kernel<<<(NE * 8 + 255) / 256, 256, 0, stream>>>(row, col, rep, x6, y, NE);

    // --- out[i] = x6[i].y[i] + x1[i,0] + y[i].lin2_w ---
    final_kernel<<<(NN + 255) / 256, 256, 0, stream>>>(x1, x6, y, deg, lin2w, out, NN);
}
